// SLM_9603546874252
// MI455X (gfx1250) — matrix operand order repaired and checked by reading, not compiled, not run on hardware
//
#include <hip/hip_runtime.h>

typedef __attribute__((ext_vector_type(8)))  float  v8f;
typedef __attribute__((ext_vector_type(8)))  __bf16 v8bf;
typedef __attribute__((ext_vector_type(16))) __bf16 v16bf;

#define N_NODES     1024
#define N_TRIALS    32
#define N_INPUTS    128
#define N_OUTPUTS   64
#define N_TIMES     1000
#define NODES_PER_WG 64
#define N_WG        (N_NODES / NODES_PER_WG)   // 16 persistent workgroups
#define THREADS     256                        // 8 wave32 per WG
#define G_CONST     0.5f
#define H_CONST     0.01f
#define ETA_CONST   1.0e-4f                    // 0.001*sqrt(0.01)
#define ZPLANE      (N_TRIALS * N_NODES)       // 32768 elements
#define REC_FLOATS  ((size_t)N_TRIALS * N_TIMES * N_NODES * 2)

// ---------------- deterministic counter-based RNG (PCG + Box-Muller) --------
__device__ __forceinline__ unsigned pcg_hash(unsigned x) {
    x = x * 747796405u + 2891336453u;
    x = ((x >> ((x >> 28u) + 4u)) ^ x) * 277803737u;
    return (x >> 22u) ^ x;
}

__device__ __forceinline__ void gauss2(unsigned sid, float* g0, float* g1) {
    unsigned u1 = pcg_hash(sid * 2u + 1u);
    unsigned u2 = pcg_hash(sid * 2u + 2u);
    float r1 = ((float)u1 + 1.0f) * 2.3283064365386963e-10f;  // (0,1]
    float r2 = (float)u2 * 2.3283064365386963e-10f;           // [0,1)
    float rad = __fsqrt_rn(-2.0f * __logf(r1));
    float s, c;
    __sincosf(6.28318530717958647f * r2, &s, &c);
    *g0 = rad * c;
    *g1 = rad * s;
}

// ---------------- device-wide barrier (persistent-kernel grid sync) ---------
__device__ __forceinline__ void grid_sync(unsigned* bar) {
    __syncthreads();
    if (threadIdx.x == 0) {
        __threadfence();
        volatile unsigned* gen = bar + 1;
        unsigned g = *gen;
        if (atomicAdd(bar, 1u) == (unsigned)gridDim.x - 1u) {
            atomicExch(bar, 0u);
            __threadfence();
            atomicAdd((unsigned*)(bar + 1), 1u);
        } else {
            while (*gen == g) { __builtin_amdgcn_s_sleep(1); }
        }
        __threadfence();
    }
    __syncthreads();
}

// Build a 16-element bf16 fragment from two 16-byte LDS chunks (2x ds_load_b128)
__device__ __forceinline__ v16bf ld_frag(const __bf16* p0, const __bf16* p1) {
    v8bf lo = *reinterpret_cast<const v8bf*>(p0);
    v8bf hi = *reinterpret_cast<const v8bf*>(p1);
    return __builtin_shufflevector(lo, hi, 0,1,2,3,4,5,6,7,8,9,10,11,12,13,14,15);
}

// Async DMA: 16 bytes per lane, global -> LDS, tracked by ASYNCcnt (no VGPR data)
__device__ __forceinline__ void async_copy_b128(unsigned lds_addr, const void* gaddr) {
    asm volatile("global_load_async_to_lds_b128 %0, %1, off"
                 :: "v"(lds_addr), "v"(gaddr) : "memory");
}

__global__ __launch_bounds__(THREADS, 1)
void SL_net_persistent(const float* __restrict__ input,
                       const float* __restrict__ a_arr,
                       const float* __restrict__ f_arr,
                       const float* __restrict__ Wrec,
                       const float* __restrict__ Wi,
                       const float* __restrict__ bi,
                       const float* __restrict__ Wo,
                       const float* __restrict__ bo,
                       float* __restrict__ out,
                       float* __restrict__ zbuf,     // fp32 ping-pong state
                       __bf16* __restrict__ zbf,     // bf16 shadow ping-pong
                       unsigned* __restrict__ bar) {
    // 256.5 KB LDS per WG (<= 320 KB WGP limit): W slice + bf16 z mirror
    __shared__ __bf16 sW [NODES_PER_WG * N_NODES];   // 128 KB, W_rec[n,k] slice
    __shared__ __bf16 sZr[N_TRIALS * N_NODES];       //  64 KB
    __shared__ __bf16 sZi[N_TRIALS * N_NODES];       //  64 KB
    __shared__ float  sRowSum[NODES_PER_WG];
    __shared__ float  sDrive [NODES_PER_WG];

    const int tid   = threadIdx.x;
    const int wg    = blockIdx.x;
    const int node0 = wg * NODES_PER_WG;

    float*  zre    = zbuf;                 // [2][ZPLANE]
    float*  zim    = zbuf + 2 * ZPLANE;    // [2][ZPLANE]
    __bf16* zbf_re = zbf;                  // [2][ZPLANE]
    __bf16* zbf_im = zbf + 2 * ZPLANE;     // [2][ZPLANE]

    // ---- one-time: stage W_rec slice to LDS (bf16) + row sums --------------
    if (tid < NODES_PER_WG) sRowSum[tid] = 0.0f;
    __syncthreads();
    for (int nL = 0; nL < NODES_PER_WG; ++nL) {
        const float* wr = Wrec + (size_t)(node0 + nL) * N_NODES;
        float part = 0.0f;
        for (int k = tid; k < N_NODES; k += THREADS) {
            float w = wr[k];
            part += w;
            sW[nL * N_NODES + k] = (__bf16)w;
        }
        atomicAdd(&sRowSum[nL], part);
    }

    // ---- one-time: z0 = H*(randn + i randn) for this WG's node slice -------
    for (int i = tid; i < N_TRIALS * NODES_PER_WG; i += THREADS) {
        int m = i >> 6;
        int n = node0 + (i & 63);
        float g0, g1;
        gauss2((unsigned)(m * N_NODES + n), &g0, &g1);
        float z0r = H_CONST * g0, z0i = H_CONST * g1;
        size_t zo = (size_t)m * N_NODES + n;
        zre[zo] = z0r;           zim[zo] = z0i;
        zbf_re[zo] = (__bf16)z0r; zbf_im[zo] = (__bf16)z0i;
    }
    grid_sync(bar);

    // per-wave tile assignment: 2 m-tiles x 4 n-tiles = 8 tiles = 8 waves
    const int lane  = tid & 31;
    const int wave  = tid >> 5;
    const int half  = lane >> 4;          // lane group 0..15 / 16..31
    const int idx   = lane & 15;
    const int mBase = (wave >> 2) * 16;   // 0, 16
    const int nBase = (wave & 3) * 16;    // 0, 16, 32, 48
    const int nL_u  = nBase + idx;
    const int n_u   = node0 + nL_u;

    const float rs = sRowSum[nL_u];       // final after grid_sync's syncthreads
    const float an = a_arr[n_u];
    const float fn = f_arr[n_u];

    const unsigned sZr_base = (unsigned)(size_t)&sZr[0];
    const unsigned sZi_base = (unsigned)(size_t)&sZi[0];

    for (int t = 0; t < N_TIMES; ++t) {
        const float*  zreC = zre + (t & 1) * ZPLANE;
        const float*  zimC = zim + (t & 1) * ZPLANE;
        float*  zreN = zre + ((t + 1) & 1) * ZPLANE;
        float*  zimN = zim + ((t + 1) & 1) * ZPLANE;
        const __bf16* zbrC = zbf_re + (t & 1) * ZPLANE;
        const __bf16* zbiC = zbf_im + (t & 1) * ZPLANE;
        __bf16* zbrN = zbf_re + ((t + 1) & 1) * ZPLANE;
        __bf16* zbiN = zbf_im + ((t + 1) & 1) * ZPLANE;

        // stage full bf16 z state global -> LDS via async DMA (no VGPR traffic)
        // 8 bf16 = 16 B per lane per op; 32768/(256*8) = 16 uniform iterations
        for (int i = tid * 8; i < ZPLANE; i += THREADS * 8) {
            async_copy_b128(sZr_base + (unsigned)i * 2u, (const void*)(zbrC + i));
            async_copy_b128(sZi_base + (unsigned)i * 2u, (const void*)(zbiC + i));
        }

        // overlap: drive[t, n] = input[t] . Wi[n] + bi[n] for this WG's nodes
        if (tid < NODES_PER_WG) {
            const float* wir = Wi + (size_t)(node0 + tid) * N_INPUTS;
            const float* inr = input + (size_t)t * N_INPUTS;
            float s = bi[node0 + tid];
            #pragma unroll 8
            for (int k = 0; k < N_INPUTS; ++k) s = __fmaf_rn(inr[k], wir[k], s);
            sDrive[tid] = s;
        }

        asm volatile("s_wait_asynccnt 0x0" ::: "memory");
        __syncthreads();

        // ---- coupling GEMM: coup = z @ W_rec.T  (re & im share B fragment)
        v8f accR = {0.f,0.f,0.f,0.f,0.f,0.f,0.f,0.f};
        v8f accI = {0.f,0.f,0.f,0.f,0.f,0.f,0.f,0.f};
        // A fragment (16-bit 16x32): lanes 0-15 hold K {0..7,16..23}, 16-31 {8..15,24..31}
        const __bf16* aRowR = sZr + (size_t)(mBase + idx) * N_NODES;
        const __bf16* aRowI = sZi + (size_t)(mBase + idx) * N_NODES;
        // B fragment (32x16): lane = column N, same K order as A: lanes 0-15 K {0..7,16..23}, 16-31 {8..15,24..31}
        const __bf16* bRow  = sW  + (size_t)nL_u * N_NODES;
        #pragma unroll 2
        for (int k0 = 0; k0 < N_NODES; k0 += 32) {
            const int ka = k0 + half * 8;
            const int kb = k0 + half * 16;
            v16bf aR = ld_frag(aRowR + ka, aRowR + ka + 16);
            v16bf aI = ld_frag(aRowI + ka, aRowI + ka + 16);
            v16bf bW = ld_frag(bRow + ka,  bRow + ka + 16);
            accR = __builtin_amdgcn_wmma_f32_16x16x32_bf16(false, aR, false, bW,
                                                           (short)0, accR, false, false);
            accI = __builtin_amdgcn_wmma_f32_16x16x32_bf16(false, aI, false, bW,
                                                           (short)0, accI, false, false);
        }

        // ---- Euler-Maruyama update + rec emission (C/D layout: VGPR r -> M)
        const float drv = sDrive[nL_u];
        #pragma unroll
        for (int r = 0; r < 8; ++r) {
            const int m = mBase + half * 8 + r;
            const size_t zo = (size_t)m * N_NODES + n_u;
            float zr = zreC[zo], zi = zimC[zo];
            float mag2  = zr * zr + zi * zi;            // |z*z| == |z|^2
            float ar    = an - mag2;
            float ode_r = zr * ar - zi * fn;            // z*(a + i f - |z|^2)
            float ode_i = zr * fn + zi * ar;
            float cr = accR[r] - zr * rs;               // z@W^T - z*row_sum
            float ci = accI[r] - zi * rs;
            float g0, g1;
            gauss2((unsigned)(((t + 1) * N_TRIALS + m) * N_NODES + n_u), &g0, &g1);
            float nzr = zr + H_CONST * ode_r + (H_CONST * G_CONST) * cr
                           + H_CONST * drv + ETA_CONST * g0;
            float nzi = zi + H_CONST * ode_i + (H_CONST * G_CONST) * ci
                           + ETA_CONST * g1;
            zreN[zo] = nzr;
            zimN[zo] = nzi;
            zbrN[zo] = (__bf16)nzr;                     // bf16 shadow for async stage
            zbiN[zo] = (__bf16)nzi;
            float2 rc; rc.x = nzr; rc.y = nzi;          // complex64 interleave
            *reinterpret_cast<float2*>(out + (((size_t)m * N_TIMES + t) * N_NODES + n_u) * 2) = rc;
        }
        grid_sync(bar);   // writes to next-state buffers visible before next step
    }

    // ---- readout: out = zT.real @ Wo.T + bo, split across all WGs ----------
    const float* zfin = zre + (N_TIMES & 1) * ZPLANE;
    int gid = wg * THREADS + tid;
    if (gid < N_TRIALS * N_OUTPUTS) {
        int m = gid >> 6;
        int o = gid & 63;
        const float* wor  = Wo + (size_t)o * N_NODES;
        const float* zrow = zfin + (size_t)m * N_NODES;
        float s = bo[o];
        for (int k = 0; k < N_NODES; ++k) s = __fmaf_rn(zrow[k], wor[k], s);
        out[REC_FLOATS + (size_t)m * N_OUTPUTS + o] = s;
    }
}

extern "C" void kernel_launch(void* const* d_in, const int* in_sizes, int n_in,
                              void* d_out, int out_size, void* d_ws, size_t ws_size,
                              hipStream_t stream) {
    const float* input = (const float*)d_in[0];
    const float* a     = (const float*)d_in[1];
    const float* f     = (const float*)d_in[2];
    const float* Wrec  = (const float*)d_in[3];
    const float* Wi    = (const float*)d_in[4];
    const float* bi    = (const float*)d_in[5];
    const float* Wo    = (const float*)d_in[6];
    const float* bo    = (const float*)d_in[7];
    float* out = (float*)d_out;

    // workspace layout:
    //   [0,256)                      barrier counters
    //   [256, 256+512K)              fp32 z ping-pong (zre[2], zim[2])
    //   [256+512K, 256+768K)         bf16 z ping-pong shadow (re[2], im[2])
    unsigned* bar  = (unsigned*)d_ws;
    float*    zbuf = (float*)((char*)d_ws + 256);
    __bf16*   zbf  = (__bf16*)((char*)d_ws + 256 + 4 * ZPLANE * sizeof(float));
    hipMemsetAsync(d_ws, 0, 256, stream);   // reset barrier each call (capture-safe)

    SL_net_persistent<<<N_WG, THREADS, 0, stream>>>(
        input, a, f, Wrec, Wi, bi, Wo, bo, out, zbuf, zbf, bar);
}
